// MultiHeadAttention_44109314130332
// MI455X (gfx1250) — matrix operand order repaired and checked by reading, not compiled, not run on hardware
//
#include <hip/hip_runtime.h>
#include <hip/hip_bf16.h>

// ---------------------------------------------------------------------------
// MultiHeadAttention for MI455X (gfx1250, wave32, WMMA 16x16x32 bf16)
//
// B=8, C=256, P=1024, T=16, H=8, D=32, DT=D*T=512, N=P*T=16384
// SCALE = sqrt(512)
//
// Pipeline (all on `stream`, graph-capturable):
//   k0  wconv  : Wk/Wq/Wv f32 -> bf16 (row-major [o][c])
//   k1  xtrans : x[b][c][n] f32 -> xT[b][n][c] bf16 (LDS-tiled transpose)
//   k2  proj   : Y^T = xT * W^T via v_wmma_f32_16x16x32_bf16
//               K,Q stored as [bh][p][dt] bf16 ; V stored transposed
//               as VT[bh][dt][p] bf16 (so pass-2 B-frags are contiguous)
//   k3  stats  : per (b,h,p): M[p]=max_q S/SCALE, Z[p]=sum exp. Block-shared
//               K-tile staged in LDS via global_load_async_to_lds_b128.
//   k4  attnout: per (b,h,q-tile): recompute S^T tiles = Q*K^T (WMMA, Q-tile
//               async-staged in LDS), weight = exp(S/SCALE - M[p]) / Z[p],
//               bf16 attn tile via LDS, out += attn^T * V, store f32.
//
// LDS tiles padded to 520/136-element row strides: 1040B/272B rotate the
// bank mapping by 4 banks per row -> conflict-free ds_load_b128 fragments.
//
// Workspace requirement: ~270 MB (K/Q/V bf16 201MB + xT 64MB + stats/W)
// ---------------------------------------------------------------------------

typedef __attribute__((ext_vector_type(16))) __bf16 v16bf;
typedef __attribute__((ext_vector_type(8)))  float  v8f;
typedef __attribute__((ext_vector_type(4)))  unsigned int u32x4;

union ABfrag { v16bf v; u32x4 q[2]; };

#define INV_SCALE 0.04419417382415922f   // 1/sqrt(512)
#define KQ_STRIDE 520                    // padded LDS row stride (elements)
#define AT_STRIDE 136                    // padded attn-tile row stride

__device__ __forceinline__ u32x4 ld16(const void* p) {
  return *reinterpret_cast<const u32x4*>(p);
}

__device__ __forceinline__ unsigned short f2bf(float f) {
  unsigned u = __builtin_bit_cast(unsigned, f);
  unsigned r = u + 0x7FFFu + ((u >> 16) & 1u);   // round-to-nearest-even
  return (unsigned short)(r >> 16);
}

__device__ __forceinline__ v8f wmma_bf16(const ABfrag& a, const ABfrag& b, v8f c) {
  return __builtin_amdgcn_wmma_f32_16x16x32_bf16(false, a.v, false, b.v,
                                                 (short)0, c, false, false);
}

// CDNA5 async copy: 16 bytes global -> LDS per lane, tracked by ASYNCcnt.
__device__ __forceinline__ void async_copy16(unsigned lds_byte_off,
                                             const unsigned short* gptr) {
  asm volatile("global_load_async_to_lds_b128 %0, %1, off"
               :: "v"(lds_byte_off), "v"(gptr) : "memory");
}
__device__ __forceinline__ void wait_async0() {
  asm volatile("s_wait_asynccnt 0x0" ::: "memory");
}

// Stage a 16-row x 512-element bf16 tile (row stride 512 in global) into LDS
// with padded stride KQ_STRIDE. 256 threads x 4 chunks of 16B each.
__device__ __forceinline__ void stage_tile_async(unsigned lds_base_off,
                                                 const unsigned short* gbase) {
#pragma unroll
  for (int j = 0; j < 4; j++) {
    int i = threadIdx.x + j * 256;                 // 0..1023 chunk id
    int row = i >> 6, col8 = i & 63;               // 64 x 16B chunks per row
    unsigned loff = lds_base_off + (unsigned)(row * KQ_STRIDE + col8 * 8) * 2;
    async_copy16(loff, gbase + (size_t)row * 512 + col8 * 8);
  }
  wait_async0();                                   // own wave's copies done
}

// ---------------------------------------------------------------- k0: weights
__global__ void wconv_kernel(const float* __restrict__ Wk,
                             const float* __restrict__ Wq,
                             const float* __restrict__ Wv,
                             unsigned short* __restrict__ Wb) {
  int i = blockIdx.x * 256 + threadIdx.x;          // 65536 per weight
  Wb[i]           = f2bf(Wk[i]);
  Wb[65536 + i]   = f2bf(Wq[i]);
  Wb[131072 + i]  = f2bf(Wv[i]);
}

// ------------------------------------------------------------- k1: x -> xT
__global__ void xtrans_kernel(const float* __restrict__ x,
                              unsigned short* __restrict__ xT) {
  __shared__ float tile[16][17];
  int b  = blockIdx.z;
  int n0 = blockIdx.x * 16;
  int c0 = blockIdx.y * 16;
  const float* xb = x + (size_t)b * 256 * 16384;
  tile[threadIdx.y][threadIdx.x] =
      xb[(size_t)(c0 + threadIdx.y) * 16384 + n0 + threadIdx.x];
  __syncthreads();
  xT[((size_t)b * 16384 + n0 + threadIdx.y) * 256 + c0 + threadIdx.x] =
      f2bf(tile[threadIdx.x][threadIdx.y]);
}

// ------------------------------------------------------------- k2: projection
// grid (6, 1024, 8), block 256. Wave w handles o-tile = blockIdx.x*8 + w.
__global__ void __launch_bounds__(256)
proj_kernel(const unsigned short* __restrict__ xT,
            const unsigned short* __restrict__ Wb,
            unsigned short* __restrict__ Kw,
            unsigned short* __restrict__ Qw,
            unsigned short* __restrict__ Vt) {
  const int lane = threadIdx.x & 31, wv = threadIdx.x >> 5;
  const int hi = lane >> 4, l16 = lane & 15;
  const int b = blockIdx.z, p = blockIdx.y;
  const int ot = blockIdx.x * 8 + wv;              // 0..47
  const int widx = ot >> 4;                        // 0=K,1=Q,2=V
  const int o0 = (ot & 15) * 16;

  const unsigned short* W = Wb + widx * 65536;
  const unsigned short* xrow = xT + ((size_t)b * 16384 + p * 16 + l16) * 256;
  const unsigned short* wrow = W + (size_t)(o0 + l16) * 256;

  v8f acc = {};
#pragma unroll
  for (int kk = 0; kk < 256; kk += 32) {
    ABfrag a, bb;
    a.q[0]  = ld16(xrow + kk + hi * 8);
    a.q[1]  = ld16(xrow + kk + 16 + hi * 8);
    bb.q[0] = ld16(wrow + kk + hi * 8);
    bb.q[1] = ld16(wrow + kk + 16 + hi * 8);
    acc = wmma_bf16(a, bb, acc);
  }

  const int o = o0 + l16;                          // C-frag column
  const int h = o >> 5, d = o & 31;
  if (widx < 2) {
    unsigned short* dst = (widx ? Qw : Kw);
    size_t base = (((size_t)(b * 8 + h) * 1024 + p) * 512) + d * 16 + hi * 8;
    unsigned short hh[8];
#pragma unroll
    for (int r = 0; r < 8; r++) hh[r] = f2bf(acc[r]);
    *reinterpret_cast<u32x4*>(dst + base) = *reinterpret_cast<const u32x4*>(hh);
  } else {
    size_t base = ((size_t)(b * 8 + h) * 512 + d * 16 + hi * 8) * 1024 + p;
#pragma unroll
    for (int r = 0; r < 8; r++) Vt[base + (size_t)r * 1024] = f2bf(acc[r]);
  }
}

// ---------------------------------------------------------- k3: softmax stats
// grid (64 p-tiles, 8 h, 8 b), block 256. K-tile (block-invariant) is staged
// into LDS once via async-to-LDS; Q streams from global/L2.
__global__ void __launch_bounds__(256)
stats_kernel(const unsigned short* __restrict__ Kw,
             const unsigned short* __restrict__ Qw,
             float* __restrict__ Mv, float* __restrict__ Zv) {
  __shared__ unsigned short Ktile[16 * KQ_STRIDE]; // padded, ~16.6 KB
  __shared__ float sm[8][16], sz[8][16];
  const int lane = threadIdx.x & 31, wv = threadIdx.x >> 5;
  const int hi = lane >> 4, l16 = lane & 15;
  const int bh = blockIdx.z * 8 + blockIdx.y;
  const int p0 = blockIdx.x * 16;
  const unsigned short* Kbase = Kw + (size_t)bh * 1024 * 512;
  const unsigned short* Qbase = Qw + (size_t)bh * 1024 * 512;

  stage_tile_async((unsigned)(size_t)&Ktile[0], Kbase + (size_t)p0 * 512);
  __syncthreads();
  const unsigned short* krow = &Ktile[l16 * KQ_STRIDE];

  float m[8], z[8];
#pragma unroll
  for (int r = 0; r < 8; r++) { m[r] = -1e30f; z[r] = 0.0f; }

  for (int j = 0; j < 8; j++) {
    const int q0 = (j * 8 + wv) * 16;
    const unsigned short* qrow = Qbase + (size_t)(q0 + l16) * 512;
    v8f acc = {};
#pragma unroll
    for (int kk = 0; kk < 512; kk += 32) {
      ABfrag a, bb;
      a.q[0]  = ld16(krow + kk + hi * 8);          // ds_load_b128
      a.q[1]  = ld16(krow + kk + 16 + hi * 8);
      bb.q[0] = ld16(qrow + kk + hi * 8);
      bb.q[1] = ld16(qrow + kk + 16 + hi * 8);
      acc = wmma_bf16(a, bb, acc);
    }
#pragma unroll
    for (int r = 0; r < 8; r++) {                  // online max/sumexp
      float s = acc[r] * INV_SCALE;
      float nm = fmaxf(m[r], s);
      z[r] = z[r] * __expf(m[r] - nm) + __expf(s - nm);
      m[r] = nm;
    }
  }
  // combine across the 16 lanes sharing each row (xor stays in 16-group)
#pragma unroll
  for (int mask = 1; mask < 16; mask <<= 1) {
#pragma unroll
    for (int r = 0; r < 8; r++) {
      float om = __shfl_xor(m[r], mask, 32);
      float oz = __shfl_xor(z[r], mask, 32);
      float nm = fmaxf(m[r], om);
      z[r] = z[r] * __expf(m[r] - nm) + oz * __expf(om - nm);
      m[r] = nm;
    }
  }
  if (l16 == 0) {
#pragma unroll
    for (int r = 0; r < 8; r++) { sm[wv][r + 8 * hi] = m[r]; sz[wv][r + 8 * hi] = z[r]; }
  }
  __syncthreads();
  if (threadIdx.x < 16) {
    int row = threadIdx.x;
    float fm = -1e30f, fz = 0.0f;
#pragma unroll
    for (int w2 = 0; w2 < 8; w2++) {
      float om = sm[w2][row], oz = sz[w2][row];
      float nm = fmaxf(fm, om);
      fz = fz * __expf(fm - nm) + oz * __expf(om - nm);
      fm = nm;
    }
    Mv[bh * 1024 + p0 + row] = fm;
    Zv[bh * 1024 + p0 + row] = fz;
  }
}

// -------------------------------------------------------------- k4: output
// grid (64 q-tiles, 8 h, 8 b), block 256. Q-tile (block-invariant) async-
// staged in LDS. Per chunk of 128 p: each wave builds one S^T tile, weights
// it into the shared attn tile, then all waves accumulate 64 output columns.
__global__ void __launch_bounds__(256)
attnout_kernel(const unsigned short* __restrict__ Kw,
               const unsigned short* __restrict__ Qw,
               const unsigned short* __restrict__ Vt,
               const float* __restrict__ Mv, const float* __restrict__ Zv,
               float* __restrict__ out) {
  __shared__ unsigned short Qtile[16 * KQ_STRIDE]; // ~16.6 KB
  __shared__ unsigned short attn[16 * AT_STRIDE];  // [q][p-local], ~4.3 KB
  const int lane = threadIdx.x & 31, wv = threadIdx.x >> 5;
  const int hi = lane >> 4, l16 = lane & 15;
  const int b = blockIdx.z, h = blockIdx.y, bh = b * 8 + h;
  const int q0 = blockIdx.x * 16;

  const unsigned short* Kbase = Kw + (size_t)bh * 1024 * 512;
  const unsigned short* Vbase = Vt + (size_t)bh * 512 * 1024;
  const float* Mp = Mv + bh * 1024;
  const float* Zp = Zv + bh * 1024;

  stage_tile_async((unsigned)(size_t)&Qtile[0],
                   Qw + (size_t)bh * 1024 * 512 + (size_t)q0 * 512);
  __syncthreads();
  const unsigned short* qrow = &Qtile[l16 * KQ_STRIDE];

  v8f acc[4] = {{}, {}, {}, {}};

  for (int chunk = 0; chunk < 8; chunk++) {
    const int p0 = (chunk * 8 + wv) * 16;
    const unsigned short* krow = Kbase + (size_t)(p0 + l16) * 512;
    v8f s = {};
#pragma unroll
    for (int kk = 0; kk < 512; kk += 32) {         // S^T[q,p] = Q * K^T
      ABfrag a, bb;
      a.q[0]  = ld16(qrow + kk + hi * 8);          // ds_load_b128
      a.q[1]  = ld16(qrow + kk + 16 + hi * 8);
      bb.q[0] = ld16(krow + kk + hi * 8);
      bb.q[1] = ld16(krow + kk + 16 + hi * 8);
      s = wmma_bf16(a, bb, s);
    }
    const int p = p0 + l16;                        // C-frag column = p
    const float mp = Mp[p];
    const float zi = 1.0f / Zp[p];
#pragma unroll
    for (int r = 0; r < 8; r++) {
      float wgt = __expf(s[r] * INV_SCALE - mp) * zi;
      attn[(r + 8 * hi) * AT_STRIDE + wv * 16 + l16] = f2bf(wgt);
    }
    __syncthreads();
#pragma unroll
    for (int ct = 0; ct < 4; ct++) {               // out += attn^T * V
      const int col0 = wv * 64 + ct * 16;
      const unsigned short* vrow =
          Vbase + (size_t)(col0 + l16) * 1024 + chunk * 128;
#pragma unroll
      for (int ks = 0; ks < 4; ks++) {
        ABfrag a, bb;
        const unsigned short* arow = attn + l16 * AT_STRIDE + ks * 32;
        a.q[0]  = ld16(arow + hi * 8);             // ds_load_b128
        a.q[1]  = ld16(arow + 16 + hi * 8);
        bb.q[0] = ld16(vrow + ks * 32 + hi * 8);
        bb.q[1] = ld16(vrow + ks * 32 + 16 + hi * 8);
        acc[ct] = wmma_bf16(a, bb, acc[ct]);
      }
    }
    __syncthreads();
  }
  // store: col = wv*64 + ct*16 + l16 -> d = wv*4 + ct, t = l16
#pragma unroll
  for (int ct = 0; ct < 4; ct++) {
    const int d = wv * 4 + ct;
    float* obase = out + ((size_t)b * 256 + h * 32 + d) * 1024 * 16;
#pragma unroll
    for (int r = 0; r < 8; r++) {
      int q = q0 + r + 8 * hi;
      obase[(size_t)q * 16 + l16] = acc[ct][r];
    }
  }
}

// ---------------------------------------------------------------------------
extern "C" void kernel_launch(void* const* d_in, const int* in_sizes, int n_in,
                              void* d_out, int out_size, void* d_ws, size_t ws_size,
                              hipStream_t stream) {
  (void)in_sizes; (void)n_in; (void)out_size; (void)ws_size;
  const float* x  = (const float*)d_in[0];
  const float* Wk = (const float*)d_in[1];
  const float* Wq = (const float*)d_in[2];
  const float* Wv = (const float*)d_in[3];

  // workspace layout (~270 MB total)
  char* ws = (char*)d_ws;
  unsigned short* Kw = (unsigned short*)ws;                    // 64 MiB
  unsigned short* Qw = Kw + (size_t)33554432;                  // 64 MiB
  unsigned short* Vt = Qw + (size_t)33554432;                  // 64 MiB
  float* Mv = (float*)(ws + 3ull * 67108864);                  // 256 KiB
  float* Zv = Mv + 65536;                                      // 256 KiB
  unsigned short* Wb = (unsigned short*)(Zv + 65536);          // 384 KiB
  unsigned short* xT = Wb + 3 * 65536;                         // 64 MiB

  wconv_kernel<<<256, 256, 0, stream>>>(Wk, Wq, Wv, Wb);
  xtrans_kernel<<<dim3(1024, 16, 8), dim3(16, 16), 0, stream>>>(x, xT);
  proj_kernel<<<dim3(6, 1024, 8), 256, 0, stream>>>(xT, Wb, Kw, Qw, Vt);
  stats_kernel<<<dim3(64, 8, 8), 256, 0, stream>>>(Kw, Qw, Mv, Zv);
  attnout_kernel<<<dim3(64, 8, 8), 256, 0, stream>>>(Kw, Qw, Vt, Mv, Zv,
                                                     (float*)d_out);
}
